// Attention_9998683865766
// MI455X (gfx1250) — matrix operand order repaired and checked by reading, not compiled, not run on hardware
//
#include <hip/hip_runtime.h>
#include <hip/hip_bf16.h>

#define DIM     768
#define NHEAD   12
#define HDIM    64
#define BATCH   4
#define SEQ     2048
#define MROWS   (BATCH*SEQ)     // 8192
#define QKV_OUT (3*DIM)         // 2304
#define SM_SCALE 0.125f         // 1/sqrt(64)

typedef __attribute__((ext_vector_type(16))) _Float16 v16h;
typedef __attribute__((ext_vector_type(8)))  _Float16 v8h;
typedef __attribute__((ext_vector_type(8)))  float    v8f;

__device__ __forceinline__ v8f wmma_f16(v16h a, v16h b, v8f c) {
  // D = A(16x32 f16) * B(32x16 f16) + C(16x16 f32)
  return __builtin_amdgcn_wmma_f32_16x16x32_f16(false, a, false, b, (short)0, c,
                                                false, false);
}

// ---- CDNA5 async global->LDS copy (ASYNCcnt-tracked), per CDNA5 ISA ----
__device__ __forceinline__ unsigned lds_off(const void* p) {
  // flat address of an LDS object: low 32 bits == workgroup-relative LDS offset
  return (unsigned)(unsigned long long)p;
}
__device__ __forceinline__ void async_copy_b128(unsigned dst_lds,
                                                const void* gsrc) {
  asm volatile("global_load_async_to_lds_b128 %0, %1, off"
               :: "v"(dst_lds), "v"(gsrc)
               : "memory");
}
__device__ __forceinline__ void wait_async_le3() {
  asm volatile("s_wait_asynccnt 0x3" ::: "memory");
}
__device__ __forceinline__ void wait_async_le4() {
  asm volatile("s_wait_asynccnt 0x4" ::: "memory");
}
__device__ __forceinline__ void wait_async_0() {
  asm volatile("s_wait_asynccnt 0x0" ::: "memory");
}

// ---------------------------------------------------------------- casts
__global__ void cast_f32_to_f16(const float* __restrict__ src,
                                _Float16* __restrict__ dst, int n) {
  int i = blockIdx.x * blockDim.x + threadIdx.x;
  int stride = gridDim.x * blockDim.x;
  for (; i < n; i += stride) dst[i] = (_Float16)src[i];
}

// ---------------------------------------------------------------- GEMM core
// 128(m) x 64(n) block tile, BK=32, 8 waves each owning a 32x32 output block
// (2x2 accumulators -> 4 wmma per K-step from 2 A-frags + 2 B-frags).
template <typename EPI>
__device__ __forceinline__ void gemm_core(const _Float16* __restrict__ Ag,
                                          const _Float16* __restrict__ Wg,
                                          EPI epilogue) {
  __shared__ __align__(16) _Float16 At[2][128][32];  // [m][k]  16KB
  __shared__ __align__(16) _Float16 Bt[2][64][32];   // [oc][k]  8KB
  const int tid  = threadIdx.x;
  const int lane = tid & 31;
  const int w    = tid >> 5;
  const int wm   = w & 3, wn = w >> 2;               // 32x32 block per wave
  const int m0   = blockIdx.x * 128;
  const int n0   = blockIdx.y * 64;
  const int lm   = lane & 15;
  const int lh   = lane >> 4;
  v8f acc00 = {}, acc01 = {}, acc10 = {}, acc11 = {};

  const int ixb = tid * 8;
  const int bm = ixb >> 5, bk = ixb & 31;
  const int NIT = DIM / 32;                          // 24

  // prologue: stage tile 0 (3 async b128 per thread per tile)
#pragma unroll
  for (int j = 0; j < 2; ++j) {
    int ix = (tid + j * 256) * 8;
    int am = ix >> 5, ak = ix & 31;
    async_copy_b128(lds_off(&At[0][am][ak]),
                    &Ag[(size_t)(m0 + am) * DIM + ak]);
  }
  async_copy_b128(lds_off(&Bt[0][bm][bk]),
                  &Wg[(size_t)(n0 + bm) * DIM + bk]);

  for (int it = 0; it < NIT; ++it) {
    const int c = it & 1;
    if (it > 0) __syncthreads();       // WAR: all done reading buf 1-c
    if (it + 1 < NIT) {
      const int kn = (it + 1) * 32;
#pragma unroll
      for (int j = 0; j < 2; ++j) {
        int ix = (tid + j * 256) * 8;
        int am = ix >> 5, ak = ix & 31;
        async_copy_b128(lds_off(&At[1 - c][am][ak]),
                        &Ag[(size_t)(m0 + am) * DIM + kn + ak]);
      }
      async_copy_b128(lds_off(&Bt[1 - c][bm][bk]),
                      &Wg[(size_t)(n0 + bm) * DIM + kn + bk]);
      wait_async_le3();                // in-order: tile `it` has landed
    } else {
      wait_async_0();
    }
    __syncthreads();                   // tile `it` visible to all waves

    v16h a0, a1;
    *(v8h*)&a0       = *(const v8h*)&At[c][wm * 32 + lm][lh * 8];
    *((v8h*)&a0 + 1) = *(const v8h*)&At[c][wm * 32 + lm][lh * 8 + 16];
    *(v8h*)&a1       = *(const v8h*)&At[c][wm * 32 + 16 + lm][lh * 8];
    *((v8h*)&a1 + 1) = *(const v8h*)&At[c][wm * 32 + 16 + lm][lh * 8 + 16];
    v16h b0, b1;
    *(v8h*)&b0       = *(const v8h*)&Bt[c][wn * 32 + lm][lh * 8];
    *((v8h*)&b0 + 1) = *(const v8h*)&Bt[c][wn * 32 + lm][lh * 8 + 16];
    *(v8h*)&b1       = *(const v8h*)&Bt[c][wn * 32 + 16 + lm][lh * 8];
    *((v8h*)&b1 + 1) = *(const v8h*)&Bt[c][wn * 32 + 16 + lm][lh * 8 + 16];
    acc00 = wmma_f16(a0, b0, acc00);
    acc01 = wmma_f16(a0, b1, acc01);
    acc10 = wmma_f16(a1, b0, acc10);
    acc11 = wmma_f16(a1, b1, acc11);
  }

  const int rbase = lh * 8;
#pragma unroll
  for (int i = 0; i < 2; ++i) {
#pragma unroll
    for (int j = 0; j < 2; ++j) {
      v8f acc = (i == 0) ? (j == 0 ? acc00 : acc01)
                         : (j == 0 ? acc10 : acc11);
      int oc = n0 + wn * 32 + j * 16 + lm;
      int mrow = m0 + wm * 32 + i * 16 + rbase;
      epilogue(acc, mrow, oc);
    }
  }
}

// ---------------------------------------------------------------- QKV GEMM
__global__ void __launch_bounds__(256)
gemm_qkv(const _Float16* __restrict__ Xh, const _Float16* __restrict__ Wh,
         _Float16* __restrict__ Qb, _Float16* __restrict__ Kb,
         _Float16* __restrict__ Vb) {
  gemm_core(Xh, Wh, [=](v8f acc, int mrow, int oc) {
    int three = oc / DIM;
    int rem   = oc % DIM;
    int h     = rem / HDIM, d = rem % HDIM;
    _Float16* dst = (three == 0) ? Qb : (three == 1) ? Kb : Vb;
#pragma unroll
    for (int r = 0; r < 8; ++r) {
      int grow = mrow + r;                           // = b*SEQ + n
      int b = grow >> 11, n = grow & (SEQ - 1);
      dst[(((size_t)(b * NHEAD + h)) * SEQ + n) * HDIM + d] = (_Float16)acc[r];
    }
  });
}

// ---------------------------------------------------------------- output proj
__global__ void __launch_bounds__(256)
gemm_proj(const _Float16* __restrict__ Ah, const _Float16* __restrict__ Wh,
          const float* __restrict__ bias, float* __restrict__ out) {
  gemm_core(Ah, Wh, [=](v8f acc, int mrow, int oc) {
    float bv = bias[oc];
#pragma unroll
    for (int r = 0; r < 8; ++r)
      out[(size_t)(mrow + r) * DIM + oc] = acc[r] + bv;
  });
}

// ---------------------------------------------------------------- flash attention
// One block per (b*h, 128-row q tile). 8 waves own 16 q rows each.
// Key tile = 64: 16 wmma per barrier epoch (8 for S, 8 for P@V).
__global__ void __launch_bounds__(256)
flash_attn(const _Float16* __restrict__ Qg, const _Float16* __restrict__ Kg,
           const _Float16* __restrict__ Vg, _Float16* __restrict__ AO) {
  __shared__ __align__(16) _Float16 Kt[2][64][64];    // K tile [k][d]   16KB
  __shared__ __align__(16) _Float16 Vraw[2][64][64];  // V row-major     16KB
  __shared__ __align__(16) _Float16 Vt[2][64][64];    // V^T [d][k]      16KB
  __shared__ __align__(16) float    Ssc[8][16][64];   // per-wave scores 32KB
  __shared__ __align__(16) _Float16 Psc[8][16][64];   // per-wave probs  16KB
  __shared__ float AlphaS[8][16];
  __shared__ float LinvS[8][16];

  const int tid = threadIdx.x, lane = tid & 31, w = tid >> 5;
  const int bh = blockIdx.x;                          // b*NHEAD + h
  const int b  = bh / NHEAD, h = bh % NHEAD;
  const int q0 = blockIdx.y * 128 + w * 16;
  const int lm = lane & 15, lh = lane >> 4, rbase = lh * 8;
  const size_t base = (size_t)bh * SEQ * HDIM;

  // Q fragments for D-chunks 0..31 and 32..63 (A operand layout)
  v16h qf[2];
#pragma unroll
  for (int c = 0; c < 2; ++c) {
    const _Float16* qp = &Qg[base + (size_t)(q0 + lm) * HDIM + c * 32 + lh * 16];
    *(v8h*)&qf[c]       = *(const v8h*)qp;
    *((v8h*)&qf[c] + 1) = *(const v8h*)(qp + 8);
  }

  v8f o0 = {}, o1 = {}, o2 = {}, o3 = {};
  float m_i = -1e30f, l_i = 0.f;

  const int NIT = SEQ / 64;                           // 32

  // prologue: stage key/value tile 0 (4 async b128 per thread per tile)
#pragma unroll
  for (int j = 0; j < 2; ++j) {
    int ix = (tid + j * 256) * 8;
    int kkj = ix >> 6, ddj = ix & 63;
    async_copy_b128(lds_off(&Kt[0][kkj][ddj]),
                    &Kg[base + (size_t)kkj * HDIM + ddj]);
    async_copy_b128(lds_off(&Vraw[0][kkj][ddj]),
                    &Vg[base + (size_t)kkj * HDIM + ddj]);
  }

  for (int it = 0; it < NIT; ++it) {
    const int c = it & 1;
    if (it > 0) __syncthreads();       // WAR window close for buf 1-c
    if (it + 1 < NIT) {
      const size_t knext = base + (size_t)(it + 1) * 64 * HDIM;
#pragma unroll
      for (int j = 0; j < 2; ++j) {
        int ix = (tid + j * 256) * 8;
        int kkj = ix >> 6, ddj = ix & 63;
        async_copy_b128(lds_off(&Kt[1 - c][kkj][ddj]),
                        &Kg[knext + (size_t)kkj * HDIM + ddj]);
        async_copy_b128(lds_off(&Vraw[1 - c][kkj][ddj]),
                        &Vg[knext + (size_t)kkj * HDIM + ddj]);
      }
      wait_async_le4();                // tile `it` (previous 4 ops) landed
    } else {
      wait_async_0();
    }
    __syncthreads();

    { // LDS->LDS transpose of V tile for the P@V B operand
#pragma unroll
      for (int j = 0; j < 2; ++j) {
        int ix = (tid + j * 256) * 8;
        int kkj = ix >> 6, ddj = ix & 63;
        v8h vv = *(const v8h*)&Vraw[c][kkj][ddj];
#pragma unroll
        for (int t = 0; t < 8; ++t) Vt[c][ddj + t][kkj] = vv[t];
      }
    }
    __syncthreads();

    // S(16q x 64k) = Q @ K^T : four key subtiles, two d-chunks
    v8f s0 = {}, s1 = {}, s2 = {}, s3 = {};
#pragma unroll
    for (int cc = 0; cc < 2; ++cc) {
      v16h bk0 = *(const v16h*)&Kt[c][0 * 16 + lm][cc * 32 + lh * 16];
      v16h bk1 = *(const v16h*)&Kt[c][1 * 16 + lm][cc * 32 + lh * 16];
      v16h bk2 = *(const v16h*)&Kt[c][2 * 16 + lm][cc * 32 + lh * 16];
      v16h bk3 = *(const v16h*)&Kt[c][3 * 16 + lm][cc * 32 + lh * 16];
      s0 = wmma_f16(qf[cc], bk0, s0);
      s1 = wmma_f16(qf[cc], bk1, s1);
      s2 = wmma_f16(qf[cc], bk2, s2);
      s3 = wmma_f16(qf[cc], bk3, s3);
    }
    // C-layout -> row-layout via per-wave LDS
#pragma unroll
    for (int r = 0; r < 8; ++r) {
      Ssc[w][rbase + r][0 * 16 + lm] = s0[r];
      Ssc[w][rbase + r][1 * 16 + lm] = s1[r];
      Ssc[w][rbase + r][2 * 16 + lm] = s2[r];
      Ssc[w][rbase + r][3 * 16 + lm] = s3[r];
    }
    // online softmax: lanes 0..15 each own one q row
    if (lane < 16) {
      float mx = m_i;
      for (int j = 0; j < 64; ++j)
        mx = fmaxf(mx, Ssc[w][lane][j] * SM_SCALE);
      float alpha = __expf(m_i - mx);
      float sum = 0.f;
      for (int j = 0; j < 64; ++j) {
        float p = __expf(Ssc[w][lane][j] * SM_SCALE - mx);
        sum += p;
        Psc[w][lane][j] = (_Float16)p;
      }
      l_i = l_i * alpha + sum;
      m_i = mx;
      AlphaS[w][lane] = alpha;
    }
    // rescale O accumulators
    float ar[8];
#pragma unroll
    for (int r = 0; r < 8; ++r) ar[r] = AlphaS[w][rbase + r];
#pragma unroll
    for (int r = 0; r < 8; ++r) {
      o0[r] *= ar[r]; o1[r] *= ar[r]; o2[r] *= ar[r]; o3[r] *= ar[r];
    }
    // O(16q x 64d) += P(16x64) @ V(64x64): two K-chunks, four d-tiles
#pragma unroll
    for (int cc = 0; cc < 2; ++cc) {
      v16h pf;
      *(v8h*)&pf       = *(const v8h*)&Psc[w][lm][cc * 32 + lh * 16];
      *((v8h*)&pf + 1) = *(const v8h*)&Psc[w][lm][cc * 32 + lh * 16 + 8];
      v16h vf0 = *(const v16h*)&Vt[c][0 * 16 + lm][cc * 32 + lh * 16];
      v16h vf1 = *(const v16h*)&Vt[c][1 * 16 + lm][cc * 32 + lh * 16];
      v16h vf2 = *(const v16h*)&Vt[c][2 * 16 + lm][cc * 32 + lh * 16];
      v16h vf3 = *(const v16h*)&Vt[c][3 * 16 + lm][cc * 32 + lh * 16];
      o0 = wmma_f16(pf, vf0, o0);
      o1 = wmma_f16(pf, vf1, o1);
      o2 = wmma_f16(pf, vf2, o2);
      o3 = wmma_f16(pf, vf3, o3);
    }
  }

  if (lane < 16) LinvS[w][lane] = 1.f / l_i;
  float li[8];
#pragma unroll
  for (int r = 0; r < 8; ++r) li[r] = LinvS[w][rbase + r];

  // write normalized O as f16 into AO [B*SEQ][DIM]
#pragma unroll
  for (int t = 0; t < 4; ++t) {
    v8f oo = (t == 0) ? o0 : (t == 1) ? o1 : (t == 2) ? o2 : o3;
    int d = t * 16 + lm;
#pragma unroll
    for (int r = 0; r < 8; ++r) {
      int qrow = q0 + rbase + r;
      AO[((size_t)(b * SEQ + qrow)) * DIM + h * HDIM + d] =
          (_Float16)(oo[r] * li[r]);
    }
  }
}

// ---------------------------------------------------------------- launch
extern "C" void kernel_launch(void* const* d_in, const int* in_sizes, int n_in,
                              void* d_out, int out_size, void* d_ws,
                              size_t ws_size, hipStream_t stream) {
  const float* x      = (const float*)d_in[0];
  const float* w_qkv  = (const float*)d_in[1];
  const float* w_proj = (const float*)d_in[2];
  const float* b_proj = (const float*)d_in[3];
  float* out = (float*)d_out;

  char* ws = (char*)d_ws;
  size_t off = 0;
  auto alloc = [&](size_t bytes) -> char* {
    char* p = ws + off;
    off += (bytes + 255) & ~(size_t)255;
    return p;
  };
  _Float16* xh  = (_Float16*)alloc((size_t)MROWS * DIM * 2);
  _Float16* wqh = (_Float16*)alloc((size_t)QKV_OUT * DIM * 2);
  _Float16* wph = (_Float16*)alloc((size_t)DIM * DIM * 2);
  _Float16* Qb  = (_Float16*)alloc((size_t)MROWS * DIM * 2);
  _Float16* Kb  = (_Float16*)alloc((size_t)MROWS * DIM * 2);
  _Float16* Vb  = (_Float16*)alloc((size_t)MROWS * DIM * 2);
  _Float16* AO  = (_Float16*)alloc((size_t)MROWS * DIM * 2);

  cast_f32_to_f16<<<dim3(512), dim3(256), 0, stream>>>(x, xh, MROWS * DIM);
  cast_f32_to_f16<<<dim3(512), dim3(256), 0, stream>>>(w_qkv, wqh,
                                                       QKV_OUT * DIM);
  cast_f32_to_f16<<<dim3(256), dim3(256), 0, stream>>>(w_proj, wph, DIM * DIM);

  gemm_qkv<<<dim3(MROWS / 128, QKV_OUT / 64), dim3(256), 0, stream>>>(
      xh, wqh, Qb, Kb, Vb);
  flash_attn<<<dim3(BATCH * NHEAD, SEQ / 128), dim3(256), 0, stream>>>(
      Qb, Kb, Vb, AO);
  gemm_proj<<<dim3(MROWS / 128, DIM / 64), dim3(256), 0, stream>>>(
      AO, wph, b_proj, out);
}
